// Attention_65386582114507
// MI455X (gfx1250) — compile-verified, matrix operand order checked by reading, not run on hardware
//
#include <hip/hip_runtime.h>

typedef __bf16 bf16;
typedef __attribute__((ext_vector_type(16))) __bf16 bf16x16;
typedef __attribute__((ext_vector_type(8)))  __bf16 bf16x8;
typedef __attribute__((ext_vector_type(4)))  __bf16 bf16x4;
typedef __attribute__((ext_vector_type(8)))  float  f32x8;

#define S_LEN 2048
#define NX    1024
#define HEADS 16
#define DHEAD 64
#define NEGV  -100000.0f

// ---------------------------------------------------------------------------
// WMMA helpers. CDNA5 16x16x32 bf16 -> f32.
// A fragment (16x32, lane = M row, h = lane>>4): element e<8 -> K = h*8+e,
// e>=8 -> K = 16+h*8+(e-8). A fragment is two contiguous 16-byte chunks at
// K-offsets h*8 and 16+h*8 when rows are stored K-contiguous. B uses the
// symmetric layout with lane = N, so B rows are kept [n][k] K-contiguous too.
// ---------------------------------------------------------------------------
static __device__ __forceinline__ bf16x16 load_frag(const bf16* __restrict__ p,
                                                    int ld, int row, int h) {
  const bf16* q = p + (size_t)row * ld + h * 8;
  bf16x8 lo = *(const bf16x8*)q;
  bf16x8 hi = *(const bf16x8*)(q + 16);
  bf16x16 f;
#pragma unroll
  for (int i = 0; i < 8; ++i) { f[i] = lo[i]; f[i + 8] = hi[i]; }
  return f;
}

static __device__ __forceinline__ f32x8 wmma_bf16f32(bf16x16 a, bf16x16 b, f32x8 c) {
  return __builtin_amdgcn_wmma_f32_16x16x32_bf16(
      /*neg_a=*/false, a, /*neg_b=*/false, b,
      /*c_mod=*/(short)0, c, /*reuse_a=*/false, /*reuse_b=*/false);
}

static __device__ __forceinline__ float rowmax16(float v) {
#pragma unroll
  for (int m = 1; m < 16; m <<= 1) v = fmaxf(v, __shfl_xor(v, m, 32));
  return v;
}
static __device__ __forceinline__ float rowsum16(float v) {
#pragma unroll
  for (int m = 1; m < 16; m <<= 1) v += __shfl_xor(v, m, 32);
  return v;
}

// ---------------------------------------------------------------------------
// Double-buffered 128x128 (BK=32) bf16-WMMA mainloop. 256 threads = 8 waves
// in a 4x2 grid; each wave owns a 32x64 patch = 2x4 WMMA accumulators.
// Staging is split into load-phase (all global loads issue back-to-back into
// registers) and store-phase (cvt_pk + vector ds stores), pipelined so the
// next tile's global loads are in flight during the WMMA burst.
// ---------------------------------------------------------------------------
static __device__ __forceinline__ void stage_gload(
    const float* __restrict__ A, const float* __restrict__ W,
    int K, int ldw, int m0, int n0, int k0, int tid,
    float4 ra[4], float rb[4][4]) {
#pragma unroll
  for (int i = 0; i < 4; ++i) {           // A tile: 128 rows x 32 k, b128 loads
    int idx = tid + i * 256;
    int row = idx >> 3, c4 = (idx & 7) * 4;
    ra[i] = *(const float4*)(A + (size_t)(m0 + row) * K + k0 + c4);
  }
  const int n  = tid & 127;
  const int kq = tid >> 7;                 // 0..1
#pragma unroll
  for (int i = 0; i < 4; ++i) {           // W tile column loads (coalesced in n)
    int kk = (i * 2 + kq) * 4;
#pragma unroll
    for (int e = 0; e < 4; ++e)
      rb[i][e] = W[(size_t)(k0 + kk + e) * ldw + n0 + n];
  }
}

static __device__ __forceinline__ void stage_sstore(
    bf16* sA, bf16* sB, int tid, const float4 ra[4], const float rb[4][4]) {
#pragma unroll
  for (int i = 0; i < 4; ++i) {
    int idx = tid + i * 256;
    int row = idx >> 3, c4 = (idx & 7) * 4;
    bf16x4 b4;
    b4[0] = (bf16)ra[i].x; b4[1] = (bf16)ra[i].y;
    b4[2] = (bf16)ra[i].z; b4[3] = (bf16)ra[i].w;
    *(bf16x4*)(sA + row * 32 + c4) = b4;
  }
  const int n  = tid & 127;
  const int kq = tid >> 7;
#pragma unroll
  for (int i = 0; i < 4; ++i) {           // transposed [n][k] b64 stores
    int kk = (i * 2 + kq) * 4;
    bf16x4 b4;
    b4[0] = (bf16)rb[i][0]; b4[1] = (bf16)rb[i][1];
    b4[2] = (bf16)rb[i][2]; b4[3] = (bf16)rb[i][3];
    *(bf16x4*)(sB + n * 32 + kk) = b4;
  }
}

static __device__ __forceinline__ void gemm128x128(
    const float* __restrict__ A, const float* __restrict__ W,
    int K, int ldw, int m0, int n0,
    bf16* sA0, bf16* sA1, bf16* sB0, bf16* sB1, f32x8 acc[2][4]) {
  const int tid  = threadIdx.x;
  const int lane = tid & 31;
  const int wid  = tid >> 5;
  const int wm   = wid & 3, wn = wid >> 2;
  const int half = lane >> 4, ln = lane & 15;
  const int NK   = K / 32;

  float4 ra[4];
  float  rb[4][4];
  stage_gload(A, W, K, ldw, m0, n0, 0, tid, ra, rb);
  stage_sstore(sA0, sB0, tid, ra, rb);

  for (int kt = 0; kt < NK; ++kt) {
    __syncthreads();
    bf16* cA = (kt & 1) ? sA1 : sA0;
    bf16* cB = (kt & 1) ? sB1 : sB0;
    bf16* nA = (kt & 1) ? sA0 : sA1;
    bf16* nB = (kt & 1) ? sB0 : sB1;
    const bool more = (kt + 1) < NK;
    if (more)  // issue next tile's global loads before the WMMA burst
      stage_gload(A, W, K, ldw, m0, n0, (kt + 1) * 32, tid, ra, rb);
    if (kt + 2 < NK) {  // global_prefetch_b8 two tiles ahead
      __builtin_prefetch(A + (size_t)(m0 + (tid >> 3)) * K + (kt + 2) * 32 + (tid & 7) * 4, 0, 1);
      __builtin_prefetch(W + (size_t)((kt + 2) * 32 + (tid >> 7) * 4) * ldw + n0 + (tid & 127), 0, 1);
    }

    bf16x16 af0 = load_frag(cA, 32, wm * 32 + ln, half);
    bf16x16 af1 = load_frag(cA, 32, wm * 32 + 16 + ln, half);
#pragma unroll
    for (int j = 0; j < 4; ++j) {
      bf16x16 bfj = load_frag(cB, 32, wn * 64 + j * 16 + ln, half);
      acc[0][j] = wmma_bf16f32(af0, bfj, acc[0][j]);
      acc[1][j] = wmma_bf16f32(af1, bfj, acc[1][j]);
    }
    if (more) stage_sstore(nA, nB, tid, ra, rb);  // fill ping-pong buffer
  }
}

// ---------------------------------------------------------------------------
// Kernel 1: qkv = x @ c_attn_w + b. Writes Q,K bf16 [B,H,S,D] and V bf16
// pre-transposed [B,H,D,S] (so attention's PV B-fragments are contiguous).
// ---------------------------------------------------------------------------
__global__ __launch_bounds__(256) void attn_qkv_gemm(
    const float* __restrict__ X, const float* __restrict__ W,
    const float* __restrict__ bias,
    bf16* __restrict__ qb, bf16* __restrict__ kb, bf16* __restrict__ vt) {
  __shared__ __align__(16) bf16 sA[2][128 * 32];
  __shared__ __align__(16) bf16 sB[2][128 * 32];
  f32x8 acc[2][4];
  f32x8 z = {};
#pragma unroll
  for (int i = 0; i < 2; ++i)
#pragma unroll
    for (int j = 0; j < 4; ++j) acc[i][j] = z;

  const int m0 = blockIdx.x * 128, n0 = blockIdx.y * 128;
  gemm128x128(X, W, NX, 3 * NX, m0, n0, sA[0], sA[1], sB[0], sB[1], acc);

  const int lane = threadIdx.x & 31;
  const int wid  = threadIdx.x >> 5;
  const int wm   = wid & 3, wn = wid >> 2;
  const int half = lane >> 4, ln = lane & 15;

#pragma unroll
  for (int im = 0; im < 2; ++im) {
#pragma unroll
    for (int j = 0; j < 4; ++j) {
      const int mb = m0 + wm * 32 + im * 16;
      const int nb = n0 + wn * 64 + j * 16;
      const int nglob = nb + ln;
      const float bv = bias[nglob];
      const int sec  = nglob >> 10;        // 0=Q 1=K 2=V
      const int nloc = nglob & (NX - 1);
      const int h = nloc >> 6, d = nloc & 63;
      const int b = mb >> 11;
      const int sbase = (mb & (S_LEN - 1)) + half * 8;
      const size_t bh = (size_t)(b * HEADS + h);
      const f32x8 a = acc[im][j];
      if (sec == 0) {
#pragma unroll
        for (int r = 0; r < 8; ++r)
          qb[(bh * S_LEN + sbase + r) * DHEAD + d] = (bf16)(a[r] + bv);
      } else if (sec == 1) {
#pragma unroll
        for (int r = 0; r < 8; ++r)
          kb[(bh * S_LEN + sbase + r) * DHEAD + d] = (bf16)(a[r] + bv);
      } else {
        bf16x8 pk;  // rows are consecutive s -> one 16B store, pre-transposed
#pragma unroll
        for (int r = 0; r < 8; ++r) pk[r] = (bf16)(a[r] + bv);
        *(bf16x8*)(vt + (bh * DHEAD + d) * S_LEN + sbase) = pk;
      }
    }
  }
}

// ---------------------------------------------------------------------------
// Kernel 2: causal flash attention. One wave per 16-row q-tile, key tiles of
// 32. QK^T and PV both on v_wmma_f32_16x16x32_bf16; online softmax in f32.
// V fragments are prefetched before the softmax VALU block to hide latency.
// ---------------------------------------------------------------------------
__global__ __launch_bounds__(128) void attn_flash(
    const bf16* __restrict__ qb, const bf16* __restrict__ kb,
    const bf16* __restrict__ vt, float* __restrict__ abuf) {
  __shared__ __align__(16) bf16 ldsP[4][16 * 32];

  const int tid  = threadIdx.x;
  const int lane = tid & 31;
  const int wid  = tid >> 5;
  const int half = lane >> 4, ln = lane & 15;
  const int bh = blockIdx.y;                 // b*16 + h
  const int q0 = blockIdx.x * 64 + wid * 16; // this wave's q rows

  const bf16* qp = qb + ((size_t)bh * S_LEN + q0) * DHEAD;
  const bf16* kp = kb + (size_t)bh * S_LEN * DHEAD;
  const bf16* vp = vt + (size_t)bh * DHEAD * S_LEN;
  bf16* pbuf = ldsP[wid];

  bf16x16 qf[2];
#pragma unroll
  for (int s = 0; s < 2; ++s) qf[s] = load_frag(qp + s * 32, DHEAD, ln, half);

  f32x8 oacc[4];
  f32x8 z = {};
#pragma unroll
  for (int t = 0; t < 4; ++t) oacc[t] = z;
  float m_r[8], l_r[8];
#pragma unroll
  for (int r = 0; r < 8; ++r) { m_r[r] = -1e30f; l_r[r] = 0.0f; }

  const int qlast = q0 + 15;
  for (int k0 = 0; k0 <= qlast; k0 += 32) {
    // S tile 16(q) x 32(key) = two 16x16 accs over two d-steps of 32
    f32x8 s0 = z, s1 = z;
#pragma unroll
    for (int s = 0; s < 2; ++s) {
      bf16x16 kf0 = load_frag(kp + (size_t)k0 * DHEAD + s * 32, DHEAD, ln, half);
      bf16x16 kf1 = load_frag(kp + (size_t)k0 * DHEAD + s * 32, DHEAD, 16 + ln, half);
      s0 = wmma_bf16f32(qf[s], kf0, s0);
      s1 = wmma_bf16f32(qf[s], kf1, s1);
    }

    // issue V^T fragment loads now; they complete under the softmax VALU work
    bf16x16 vf[4];
#pragma unroll
    for (int t = 0; t < 4; ++t)
      vf[t] = load_frag(vp + k0, S_LEN, t * 16 + ln, half);

    const bool need_mask = (k0 + 31) > q0;
    const float scl = 0.125f;  // 1/sqrt(64)
#pragma unroll
    for (int r = 0; r < 8; ++r) {
      const int qrow = q0 + half * 8 + r;
      float a = s0[r] * scl, b = s1[r] * scl;
      if (need_mask) {
        if (k0 + ln > qrow)      a = NEGV;
        if (k0 + 16 + ln > qrow) b = NEGV;
      }
      float tmax = rowmax16(fmaxf(a, b));
      float mnew = fmaxf(m_r[r], tmax);
      float corr = __expf(m_r[r] - mnew);
      float pa = __expf(a - mnew), pb = __expf(b - mnew);
      float psum = rowsum16(pa + pb);
      l_r[r] = l_r[r] * corr + psum;
      m_r[r] = mnew;
#pragma unroll
      for (int t = 0; t < 4; ++t) oacc[t][r] *= corr;
      // stage P (C/D layout) -> A layout via wave-private LDS
      pbuf[(half * 8 + r) * 32 + ln]      = (bf16)pa;
      pbuf[(half * 8 + r) * 32 + 16 + ln] = (bf16)pb;
    }
    asm volatile("s_wait_dscnt 0" ::: "memory");  // LDS in-order per wave

    bf16x16 pf = load_frag(pbuf, 32, ln, half);
#pragma unroll
    for (int t = 0; t < 4; ++t)
      oacc[t] = wmma_bf16f32(pf, vf[t], oacc[t]);
  }

  // write merged-head f32 [B,S,NX]
  const int b = bh >> 4, h = bh & 15;
#pragma unroll
  for (int t = 0; t < 4; ++t) {
    const int d = t * 16 + ln;
#pragma unroll
    for (int r = 0; r < 8; ++r) {
      const int srow = q0 + half * 8 + r;
      abuf[((size_t)b * S_LEN + srow) * NX + h * DHEAD + d] = oacc[t][r] / l_r[r];
    }
  }
}

// ---------------------------------------------------------------------------
// Kernel 3: out = abuf @ c_proj_w + c_proj_b, f32 out.
// ---------------------------------------------------------------------------
__global__ __launch_bounds__(256) void attn_proj_gemm(
    const float* __restrict__ A, const float* __restrict__ W,
    const float* __restrict__ bias, float* __restrict__ out) {
  __shared__ __align__(16) bf16 sA[2][128 * 32];
  __shared__ __align__(16) bf16 sB[2][128 * 32];
  f32x8 acc[2][4];
  f32x8 z = {};
#pragma unroll
  for (int i = 0; i < 2; ++i)
#pragma unroll
    for (int j = 0; j < 4; ++j) acc[i][j] = z;

  const int m0 = blockIdx.x * 128, n0 = blockIdx.y * 128;
  gemm128x128(A, W, NX, NX, m0, n0, sA[0], sA[1], sB[0], sB[1], acc);

  const int lane = threadIdx.x & 31;
  const int wid  = threadIdx.x >> 5;
  const int wm   = wid & 3, wn = wid >> 2;
  const int half = lane >> 4, ln = lane & 15;

#pragma unroll
  for (int im = 0; im < 2; ++im) {
#pragma unroll
    for (int j = 0; j < 4; ++j) {
      const int mb = m0 + wm * 32 + im * 16 + half * 8;
      const int nb = n0 + wn * 64 + j * 16 + ln;
      const float bv = bias[nb];
      const f32x8 a = acc[im][j];
#pragma unroll
      for (int r = 0; r < 8; ++r)
        out[(size_t)(mb + r) * NX + nb] = a[r] + bv;
    }
  }
}

// ---------------------------------------------------------------------------
// Host launcher
// ---------------------------------------------------------------------------
extern "C" void kernel_launch(void* const* d_in, const int* in_sizes, int n_in,
                              void* d_out, int out_size, void* d_ws, size_t ws_size,
                              hipStream_t stream) {
  (void)in_sizes; (void)n_in; (void)out_size; (void)ws_size;
  const float* x    = (const float*)d_in[0];
  const float* caw  = (const float*)d_in[1];  // [1024, 3072]
  const float* cab  = (const float*)d_in[2];  // [3072]
  const float* cpw  = (const float*)d_in[3];  // [1024, 1024]
  const float* cpb  = (const float*)d_in[4];  // [1024]
  float* out = (float*)d_out;

  char* ws = (char*)d_ws;
  bf16* qb   = (bf16*)(ws);                        // 8M bf16 = 16MB
  bf16* kb   = (bf16*)(ws + (size_t)16 * 1024 * 1024);
  bf16* vt   = (bf16*)(ws + (size_t)32 * 1024 * 1024);
  float* abuf = (float*)(ws + (size_t)48 * 1024 * 1024);  // 32MB

  // 1) QKV: M=8192, N=3072, K=1024
  attn_qkv_gemm<<<dim3(64, 24), 256, 0, stream>>>(x, caw, cab, qb, kb, vt);
  // 2) flash attention: (S/64) x (B*H) blocks of 4 waves
  attn_flash<<<dim3(S_LEN / 64, 4 * HEADS), 128, 0, stream>>>(qb, kb, vt, abuf);
  // 3) proj: M=8192, N=1024, K=1024
  attn_proj_gemm<<<dim3(64, 8), 256, 0, stream>>>(abuf, cpw, cpb, out);
}
